// MultiHeadAttentionBlock_28905129902443
// MI455X (gfx1250) — matrix operand order repaired and checked by reading, not compiled, not run on hardware
//
#include <hip/hip_runtime.h>
#include <hip/hip_bf16.h>

// ---------------------------------------------------------------------------
// MHA block for MI455X (gfx1250): f16 WMMA everywhere, f32 accumulate.
// B=2, N=2048, D=1024, H=16, d_h=64.  ~69 GFLOP vs ~67MB traffic: compute-
// bound -> all matmuls via v_wmma_f32_16x16x32_f16; tile staging via
// GLOBAL_LOAD_ASYNC_TO_LDS_B128 double-buffered on ASYNCcnt.
// ---------------------------------------------------------------------------

#define HD      64
#define NHEADS  16
#define DMODEL  1024
#define NSEQ    2048
#define BATCH   2
#define MROWS   (BATCH * NSEQ)   // 4096

typedef __attribute__((ext_vector_type(16))) _Float16 v16h;
typedef __attribute__((ext_vector_type(8)))  _Float16 v8h;
typedef __attribute__((ext_vector_type(8)))  float    v8f;

// ---------------------------------------------------------------------------
// WMMA wrapper: D = A(16x32 f16) * B(32x16 f16) + C(16x16 f32)
// ---------------------------------------------------------------------------
__device__ inline v8f wmma_f16(v16h a, v16h b, v8f c) {
  return __builtin_amdgcn_wmma_f32_16x16x32_f16(
      /*neg_a=*/false, a, /*neg_b=*/false, b,
      /*c_mod=*/(short)0, c, /*reuse_a=*/false, /*reuse_b=*/false);
}

__device__ inline v16h cat8(v8h lo, v8h hi) {
  return __builtin_shufflevector(lo, hi, 0, 1, 2, 3, 4, 5, 6, 7,
                                 8, 9, 10, 11, 12, 13, 14, 15);
}

// Async memory->LDS copy (ASYNCcnt-tracked, no VGPR round trip)
__device__ inline void async_load_b128(unsigned lds_off, const void* gaddr) {
  asm volatile("global_load_async_to_lds_b128 %0, %1, off"
               :: "v"(lds_off), "v"(gaddr) : "memory");
}
__device__ inline void wait_async0() {
#if __has_builtin(__builtin_amdgcn_s_wait_asynccnt)
  __builtin_amdgcn_s_wait_asynccnt(0);
#else
  asm volatile("s_wait_asynccnt 0x0" ::: "memory");
#endif
}
// Generic-pointer low 32 bits == wave-relative LDS byte address (aperture decode)
__device__ inline unsigned lds_off(const void* p) {
  return (unsigned)(uintptr_t)p;
}

// ---------------------------------------------------------------------------
// Fragment loaders (this file's contiguous-k order, not the ISA table order; wave32)
// A 16x32 (MxK): lane m=lane&15, kb=(lane>>4)*16; elems 0..15 -> K=kb..kb+15.
// B 32x16 (KxN): lane n=lane&15, kb=(lane>>4)*16; elems 0..15 -> K=kb..kb+15.
// ---------------------------------------------------------------------------
__device__ inline v16h load_frag_a(const _Float16* __restrict__ p, int ld) {
  const int lane = threadIdx.x & 31;
  const int m  = lane & 15;
  const int kb = (lane >> 4) << 4;            // 0 or 16
  v8h lo = *(const v8h*)(p + (size_t)m * ld + kb);
  v8h hi = *(const v8h*)(p + (size_t)m * ld + kb + 8);
  return cat8(lo, hi);
}

// B stored "by column": tile[n][k] row-major with row stride ld (contiguous K)
__device__ inline v16h load_frag_b(const _Float16* __restrict__ p, int ld) {
  const int lane = threadIdx.x & 31;
  const int n  = lane & 15;
  const int kb = (lane >> 4) << 4;            // 0 or 16
  v8h lo = *(const v8h*)(p + (size_t)n * ld + kb);
  v8h hi = *(const v8h*)(p + (size_t)n * ld + kb + 8);
  return cat8(lo, hi);
}

// B from a tile stored [k][n] row-major (e.g. V in LDS): strided gather
__device__ inline v16h load_frag_b_colmajor(const _Float16* __restrict__ p, int ld) {
  const int lane = threadIdx.x & 31;
  const int n  = lane & 15;
  const int kb = (lane >> 4) << 4;            // 0 or 16
  v16h out;
#pragma unroll
  for (int i = 0; i < 16; ++i) out[i] = p[(size_t)(kb + i) * ld + n];
  return out;
}

// ---------------------------------------------------------------------------
// f32 -> f16 conversion, 8 elements per thread (b128 loads, packed cvt)
// ---------------------------------------------------------------------------
__global__ void cvt_f32_to_f16x8(const float* __restrict__ x,
                                 _Float16* __restrict__ y, int n8) {
  int i = blockIdx.x * blockDim.x + threadIdx.x;
  if (i < n8) {
    v8f f = ((const v8f*)x)[i];
    v8h o;
#pragma unroll
    for (int j = 0; j < 8; ++j) o[j] = (_Float16)f[j];
    ((v8h*)y)[i] = o;
  }
}

// ---------------------------------------------------------------------------
// Projection GEMM: Y[m][c] = sum_k X[m][k] * W[c][k] + bias[c]
// M=4096, N=1024, K=1024. Block tile 64x256, 8 waves of 32x64, K-step 32.
// Double-buffered async staging of A (64x32) and B (256x32) tiles.
// OUT_MODE 0: f16 output in head-split [B,H,N,64] layout (Q/K/V projections)
// OUT_MODE 1: f32 output row-major [M,D] (final output projection)
// ---------------------------------------------------------------------------
template <int OUT_MODE>
__global__ void __launch_bounds__(256)
gemm_proj(const _Float16* __restrict__ X, const _Float16* __restrict__ W,
          const float* __restrict__ bias, _Float16* __restrict__ Yh,
          float* __restrict__ Yf) {
  __shared__ _Float16 As[2][64 * 32];
  __shared__ _Float16 Bs[2][256 * 32];

  const int tid   = threadIdx.x;
  const int wv    = tid >> 5;
  const int waveM = wv >> 2;          // 0..1
  const int waveN = wv & 3;           // 0..3
  const int mBase = blockIdx.y * 64;
  const int nBase = blockIdx.x * 256;

  v8f acc[2][4] = {};

  const int arow = tid >> 2, ach = tid & 3;   // 64 rows x 4 chunks of 8 f16
  const unsigned ldsA[2] = { lds_off(&As[0][arow * 32 + ach * 8]),
                             lds_off(&As[1][arow * 32 + ach * 8]) };
  const unsigned ldsB[2] = { lds_off(&Bs[0][arow * 32 + ach * 8]),
                             lds_off(&Bs[1][arow * 32 + ach * 8]) };
  const _Float16* gA = &X[(size_t)(mBase + arow) * DMODEL + ach * 8];
  const _Float16* gB = &W[(size_t)(nBase + arow) * DMODEL + ach * 8];

  auto stage = [&](int buf, int k0) {
    async_load_b128(ldsA[buf], gA + k0);
#pragma unroll
    for (int t = 0; t < 4; ++t)   // B rows arow, arow+64, arow+128, arow+192
      async_load_b128(ldsB[buf] + 4096u * t,
                      gB + (size_t)t * 64 * DMODEL + k0);
  };

  stage(0, 0);
  int cur = 0;
  for (int k0 = 0; k0 < DMODEL; k0 += 32) {
    wait_async0();          // this wave's tile-i loads are in LDS
    __syncthreads();        // ... and everyone else's too
    if (k0 + 32 < DMODEL) stage(cur ^ 1, k0 + 32);   // overlap next tile

    // Hoist all fragments so ds_loads can batch ahead of the WMMA chain
    v16h a0 = load_frag_a(&As[cur][(waveM * 32 + 0)  * 32], 32);
    v16h a1 = load_frag_a(&As[cur][(waveM * 32 + 16) * 32], 32);
    v16h b0 = load_frag_b(&Bs[cur][(waveN * 64 + 0)  * 32], 32);
    v16h b1 = load_frag_b(&Bs[cur][(waveN * 64 + 16) * 32], 32);
    v16h b2 = load_frag_b(&Bs[cur][(waveN * 64 + 32) * 32], 32);
    v16h b3 = load_frag_b(&Bs[cur][(waveN * 64 + 48) * 32], 32);
    acc[0][0] = wmma_f16(a0, b0, acc[0][0]);
    acc[1][0] = wmma_f16(a1, b0, acc[1][0]);
    acc[0][1] = wmma_f16(a0, b1, acc[0][1]);
    acc[1][1] = wmma_f16(a1, b1, acc[1][1]);
    acc[0][2] = wmma_f16(a0, b2, acc[0][2]);
    acc[1][2] = wmma_f16(a1, b2, acc[1][2]);
    acc[0][3] = wmma_f16(a0, b3, acc[0][3]);
    acc[1][3] = wmma_f16(a1, b3, acc[1][3]);
    cur ^= 1;
  }

  const int lane  = tid & 31;
  const int half_ = lane >> 4;
  const int ln    = lane & 15;
#pragma unroll
  for (int i = 0; i < 2; ++i) {
#pragma unroll
    for (int j = 0; j < 4; ++j) {
      const int c  = nBase + waveN * 64 + j * 16 + ln;
      const float bv = bias[c];
#pragma unroll
      for (int r = 0; r < 8; ++r) {
        const int m = mBase + waveM * 32 + i * 16 + r + 8 * half_;
        const float v = acc[i][j][r] + bv;
        if (OUT_MODE == 0) {
          const int b_ = m >> 11;             // m / NSEQ
          const int ns = m & (NSEQ - 1);
          const int h  = c >> 6, dd = c & 63;
          Yh[(((size_t)b_ * NHEADS + h) * NSEQ + ns) * HD + dd] = (_Float16)v;
        } else {
          Yf[(size_t)m * DMODEL + c] = v;
        }
      }
    }
  }
}

// ---------------------------------------------------------------------------
// Flash attention: one block per (b,h) x 128-query-row tile; 8 waves x 16 rows.
// K/V tiles double-buffered via async copies (ASYNCcnt); online softmax;
// scale folded into Q.
// ---------------------------------------------------------------------------
__global__ void __launch_bounds__(256)
attn_fa(const _Float16* __restrict__ Qp, const _Float16* __restrict__ Kp,
        const _Float16* __restrict__ Vp, _Float16* __restrict__ Oh) {
  __shared__ _Float16 Ks[2][32 * 64];
  __shared__ _Float16 Vs[2][32 * 64];
  __shared__ _Float16 Ps[8][16 * 32];

  const int bh = blockIdx.x;                 // 0..31
  const int b_ = bh >> 4, h = bh & 15;
  const int rowBase = blockIdx.y * 128;
  const int tid  = threadIdx.x;
  const int wv   = tid >> 5;
  const int lane = tid & 31;
  const int half_ = lane >> 4;
  const int ln    = lane & 15;

  const _Float16* Qh = Qp + (size_t)bh * NSEQ * HD;
  const _Float16* Kh = Kp + (size_t)bh * NSEQ * HD;
  const _Float16* Vh = Vp + (size_t)bh * NSEQ * HD;

  const int mRow = rowBase + wv * 16;
  v16h q0 = load_frag_a(Qh + (size_t)mRow * HD, HD);
  v16h q1 = load_frag_a(Qh + (size_t)mRow * HD + 32, HD);
  // softmax scale 1/sqrt(64) = 0.125 exactly representable in f16
#pragma unroll
  for (int i = 0; i < 16; ++i) {
    q0[i] = (_Float16)((float)q0[i] * 0.125f);
    q1[i] = (_Float16)((float)q1[i] * 0.125f);
  }

  float mrow[8], lrow[8];
  v8f o[4] = {};
#pragma unroll
  for (int r = 0; r < 8; ++r) { mrow[r] = -3.0e38f; lrow[r] = 0.f; }

  const int ldrow = tid >> 3;   // 0..31  (32 keys)
  const int ldch  = tid & 7;    // 0..7   (8 chunks of 8 f16 = 64 dims)
  const unsigned ldsK[2] = { lds_off(&Ks[0][ldrow * 64 + ldch * 8]),
                             lds_off(&Ks[1][ldrow * 64 + ldch * 8]) };
  const unsigned ldsV[2] = { lds_off(&Vs[0][ldrow * 64 + ldch * 8]),
                             lds_off(&Vs[1][ldrow * 64 + ldch * 8]) };
  const _Float16* gK = &Kh[(size_t)ldrow * HD + ldch * 8];
  const _Float16* gV = &Vh[(size_t)ldrow * HD + ldch * 8];

  auto stage = [&](int buf, int kb) {
    async_load_b128(ldsK[buf], gK + (size_t)kb * HD);
    async_load_b128(ldsV[buf], gV + (size_t)kb * HD);
  };

  stage(0, 0);
  int cur = 0;
  for (int kb = 0; kb < NSEQ; kb += 32) {
    wait_async0();
    __syncthreads();
    if (kb + 32 < NSEQ) stage(cur ^ 1, kb + 32);   // overlap next K/V tile

    // S = Q * K^T : hoist all K^T fragments, then run the WMMA chain
    v16h kb00 = load_frag_b(&Ks[cur][0 * 64 + 0],   64);
    v16h kb01 = load_frag_b(&Ks[cur][0 * 64 + 32],  64);
    v16h kb10 = load_frag_b(&Ks[cur][16 * 64 + 0],  64);
    v16h kb11 = load_frag_b(&Ks[cur][16 * 64 + 32], 64);
    v8f s0 = {}, s1 = {};
    s0 = wmma_f16(q0, kb00, s0);
    s1 = wmma_f16(q0, kb10, s1);
    s0 = wmma_f16(q1, kb01, s0);
    s1 = wmma_f16(q1, kb11, s1);

    // Online softmax: row values live across 16 lanes of each half-wave
    float p0[8], p1[8], alpha[8];
#pragma unroll
    for (int r = 0; r < 8; ++r) {
      float vmax = fmaxf(s0[r], s1[r]);
#pragma unroll
      for (int off = 1; off < 16; off <<= 1)
        vmax = fmaxf(vmax, __shfl_xor(vmax, off, 32));
      const float mn = fmaxf(mrow[r], vmax);
      alpha[r] = __expf(mrow[r] - mn);
      p0[r] = __expf(s0[r] - mn);
      p1[r] = __expf(s1[r] - mn);
      float s = p0[r] + p1[r];
#pragma unroll
      for (int off = 1; off < 16; off <<= 1)
        s += __shfl_xor(s, off, 32);
      lrow[r] = lrow[r] * alpha[r] + s;
      mrow[r] = mn;
    }

    // Reshape P (C-layout) -> A-fragment via wave-private LDS round trip
#pragma unroll
    for (int r = 0; r < 8; ++r) {
      Ps[wv][(r + 8 * half_) * 32 + ln]      = (_Float16)p0[r];
      Ps[wv][(r + 8 * half_) * 32 + 16 + ln] = (_Float16)p1[r];
    }
    v16h pa = load_frag_a(&Ps[wv][0], 32);

    // Hoist V fragments, rescale accumulators, then O += P * V
    v16h vb0 = load_frag_b_colmajor(&Vs[cur][0],  64);
    v16h vb1 = load_frag_b_colmajor(&Vs[cur][16], 64);
    v16h vb2 = load_frag_b_colmajor(&Vs[cur][32], 64);
    v16h vb3 = load_frag_b_colmajor(&Vs[cur][48], 64);
#pragma unroll
    for (int t = 0; t < 4; ++t)
#pragma unroll
      for (int r = 0; r < 8; ++r) o[t][r] *= alpha[r];
    o[0] = wmma_f16(pa, vb0, o[0]);
    o[1] = wmma_f16(pa, vb1, o[1]);
    o[2] = wmma_f16(pa, vb2, o[2]);
    o[3] = wmma_f16(pa, vb3, o[3]);
    cur ^= 1;
  }

  // Normalize and store to [B, N, D] f16 (merged heads) for output projection
#pragma unroll
  for (int r = 0; r < 8; ++r) {
    const float inv = 1.0f / lrow[r];
    const int mseq = mRow + r + 8 * half_;
    const size_t base = ((size_t)b_ * NSEQ + mseq) * DMODEL + h * HD;
#pragma unroll
    for (int t = 0; t < 4; ++t)
      Oh[base + t * 16 + ln] = (_Float16)(o[t][r] * inv);
  }
}

// ---------------------------------------------------------------------------
// Launcher
// ---------------------------------------------------------------------------
extern "C" void kernel_launch(void* const* d_in, const int* in_sizes, int n_in,
                              void* d_out, int out_size, void* d_ws, size_t ws_size,
                              hipStream_t stream) {
  (void)in_sizes; (void)n_in; (void)out_size; (void)ws_size;
  const float* q  = (const float*)d_in[0];
  const float* k  = (const float*)d_in[1];
  const float* v  = (const float*)d_in[2];
  const float* Wq = (const float*)d_in[3];
  const float* bq = (const float*)d_in[4];
  const float* Wk = (const float*)d_in[5];
  const float* bk = (const float*)d_in[6];
  const float* Wv = (const float*)d_in[7];
  const float* bv = (const float*)d_in[8];
  const float* Wo = (const float*)d_in[9];
  const float* bo = (const float*)d_in[10];

  const size_t ACT = (size_t)MROWS * DMODEL;     // 4096*1024
  const size_t WEL = (size_t)DMODEL * DMODEL;    // 1024*1024

  _Float16* ws  = (_Float16*)d_ws;
  _Float16* qh  = ws;  ws += ACT;
  _Float16* kh  = ws;  ws += ACT;
  _Float16* vh  = ws;  ws += ACT;
  _Float16* Wqh = ws;  ws += WEL;
  _Float16* Wkh = ws;  ws += WEL;
  _Float16* Wvh = ws;  ws += WEL;
  _Float16* Woh = ws;  ws += WEL;
  _Float16* Qp  = ws;  ws += ACT;   // [B,H,N,64]
  _Float16* Kp  = ws;  ws += ACT;
  _Float16* Vp  = ws;  ws += ACT;
  _Float16* Ohb = ws;  ws += ACT;   // [B,N,D]

  const int TB = 256;
  const int A8 = (int)(ACT / 8), W8 = (int)(WEL / 8);
  cvt_f32_to_f16x8<<<(A8 + TB - 1) / TB, TB, 0, stream>>>(q,  qh,  A8);
  cvt_f32_to_f16x8<<<(A8 + TB - 1) / TB, TB, 0, stream>>>(k,  kh,  A8);
  cvt_f32_to_f16x8<<<(A8 + TB - 1) / TB, TB, 0, stream>>>(v,  vh,  A8);
  cvt_f32_to_f16x8<<<(W8 + TB - 1) / TB, TB, 0, stream>>>(Wq, Wqh, W8);
  cvt_f32_to_f16x8<<<(W8 + TB - 1) / TB, TB, 0, stream>>>(Wk, Wkh, W8);
  cvt_f32_to_f16x8<<<(W8 + TB - 1) / TB, TB, 0, stream>>>(Wv, Wvh, W8);
  cvt_f32_to_f16x8<<<(W8 + TB - 1) / TB, TB, 0, stream>>>(Wo, Woh, W8);

  dim3 gg(DMODEL / 256, MROWS / 64);
  gemm_proj<0><<<gg, TB, 0, stream>>>(qh, Wqh, bq, Qp, nullptr);
  gemm_proj<0><<<gg, TB, 0, stream>>>(kh, Wkh, bk, Kp, nullptr);
  gemm_proj<0><<<gg, TB, 0, stream>>>(vh, Wvh, bv, Vp, nullptr);

  attn_fa<<<dim3(BATCH * NHEADS, NSEQ / 128), TB, 0, stream>>>(Qp, Kp, Vp, Ohb);

  gemm_proj<1><<<gg, TB, 0, stream>>>(Ohb, Woh, bo, nullptr, (float*)d_out);
}
